// PNANet_77884936946106
// MI455X (gfx1250) — matrix operand order repaired and checked by reading, not compiled, not run on hardware
//
#include <hip/hip_runtime.h>
#include <hip/hip_bf16.h>

// ---------------------------------------------------------------------------
// PNA-Net forward for MI455X (gfx1250).
// Edge message GEMM (dominant: ~208 of ~240 GFLOP) runs on bf16 WMMA
// (v_wmma_f32_16x16x32_bf16); A-operand gather staged to LDS with
// GLOBAL_LOAD_ASYNC_TO_LDS_B128 (ASYNCcnt); one wave per output n-tile keeps
// B-fragments register-resident; aggregation fused via native f32 L2 atomics
// (global_atomic_add_f32 / min_num_f32 / max_num_f32).
// ---------------------------------------------------------------------------

typedef __attribute__((ext_vector_type(16))) __bf16 v16bf;
typedef __attribute__((ext_vector_type(8)))  float  v8f;

#define SPAD 256          // padded stat stride per node (T*F <= 250)
#define TPB  8            // 16-edge tiles per block in the WMMA edge kernel
#define EPS_STD 1e-5f
#define EPS_BN  1e-5f

// ---------------- native float atomic min/max ------------------------------
__device__ __forceinline__ void atomicMinF(float* a, float v) {
  __hip_atomic_fetch_min(a, v, __ATOMIC_RELAXED, __HIP_MEMORY_SCOPE_AGENT);
}
__device__ __forceinline__ void atomicMaxF(float* a, float v) {
  __hip_atomic_fetch_max(a, v, __ATOMIC_RELAXED, __HIP_MEMORY_SCOPE_AGENT);
}

// ---------------- generic fill ---------------------------------------------
__global__ void fill_kernel(float* p, float v, long long count) {
  long long i = (long long)blockIdx.x * blockDim.x + threadIdx.x;
  long long stride = (long long)gridDim.x * blockDim.x;
  for (; i < count; i += stride) p[i] = v;
}

// fused stat-buffer init: sum=0, sumsq=0, min=+inf, max=-inf
__global__ void init_stats_kernel(float* ssum, float* ssq, float* smn, float* smx,
                                  long long count) {
  long long i = (long long)blockIdx.x * blockDim.x + threadIdx.x;
  long long stride = (long long)gridDim.x * blockDim.x;
  const float inf = __builtin_inff();
  for (; i < count; i += stride) {
    ssum[i] = 0.0f; ssq[i] = 0.0f; smn[i] = inf; smx[i] = -inf;
  }
}

// ---------------- degree / log-degree --------------------------------------
__global__ void deg_kernel(const int* __restrict__ dst, float* deg, int E) {
  int e = blockIdx.x * blockDim.x + threadIdx.x;
  if (e < E) atomicAdd(&deg[dst[e]], 1.0f);
}

__global__ void logd_kernel(const float* __restrict__ deg, float* logd,
                            float* sumlog, int N) {
  __shared__ float ls[256];
  float acc = 0.0f;
  for (int n = blockIdx.x * blockDim.x + threadIdx.x; n < N;
       n += gridDim.x * blockDim.x) {
    float d = deg[n];
    logd[n] = logf(1.0f + fmaxf(d, 1.0f));   // log(clamp(deg,1)+1)
    acc += logf(1.0f + d);                    // log1p(deg) for the average
  }
  ls[threadIdx.x] = acc;
  __syncthreads();
  for (int off = 128; off > 0; off >>= 1) {
    if (threadIdx.x < off) ls[threadIdx.x] += ls[threadIdx.x + off];
    __syncthreads();
  }
  if (threadIdx.x == 0) atomicAdd(sumlog, ls[0]);
}

__global__ void avg_kernel(const float* sumlog, float* avgl, int N) {
  if (blockIdx.x == 0 && threadIdx.x == 0) avgl[0] = sumlog[0] / (float)N;
}

// ---------------- weight packing -------------------------------------------
// W_pre f32 [T,2F,F] -> WpT bf16 [256][128]  (row = output column n=t*F+f,
// contiguous K: K 0..F-1 = dst half, K 64..64+F-1 = src half, zeros else)
// b_pre -> bpack f32 [256]
__global__ void pack_w(const float* __restrict__ Wpre,
                       const float* __restrict__ bpre,
                       __bf16* WpT, float* bpack, int T, int F) {
  int idx = blockIdx.x * blockDim.x + threadIdx.x;
  if (idx >= 256 * 128) return;
  int ncol = idx >> 7, k = idx & 127;
  float v = 0.0f;
  int t = ncol / F, f = ncol % F;
  if (ncol < T * F) {
    if (k < F)                     v = Wpre[((size_t)t * 2 * F + k) * F + f];
    else if (k >= 64 && k < 64 + F)v = Wpre[((size_t)t * 2 * F + F + (k - 64)) * F + f];
  }
  WpT[idx] = (__bf16)v;
  if (k == 0) bpack[ncol] = (ncol < T * F) ? bpre[(size_t)t * F + f] : 0.0f;
}

// ---------------- layer-0 scalar edge kernel (F_in=2, T=1) -----------------
__global__ void edge_small(const float* __restrict__ x,
                           const int* __restrict__ src, const int* __restrict__ dst,
                           const float* __restrict__ Wpre, const float* __restrict__ bpre,
                           float* ssum, float* ssq, float* smn, float* smx, int E) {
  int e = blockIdx.x * blockDim.x + threadIdx.x;
  if (e >= E) return;
  int s = src[e], d = dst[e];
  float h0 = x[(size_t)d * 2], h1 = x[(size_t)d * 2 + 1];
  float h2 = x[(size_t)s * 2], h3 = x[(size_t)s * 2 + 1];
#pragma unroll
  for (int f = 0; f < 2; ++f) {
    float m = bpre[f] + h0 * Wpre[0 * 2 + f] + h1 * Wpre[1 * 2 + f]
                      + h2 * Wpre[2 * 2 + f] + h3 * Wpre[3 * 2 + f];
    size_t o = (size_t)((unsigned)d * SPAD + f);
    atomicAdd(ssum + o, m);
    atomicAdd(ssq + o, m * m);
    atomicMinF(smn + o, m);
    atomicMaxF(smx + o, m);
  }
}

// ---------------- WMMA edge-message kernel (F_in=50) -----------------------
// blockDim = NT*32: one wave per 16-column n-tile, TPB*16 edges per block.
// B-fragments live in registers for the whole block; A rebuilt per tile from
// async-staged LDS.
union ABfrag { unsigned int u[8]; v16bf v; };

__global__ void __launch_bounds__(512)
edge_wmma(const __bf16* __restrict__ hbf, const __bf16* __restrict__ WpT,
          const float* __restrict__ bpack,
          const int* __restrict__ src, const int* __restrict__ dst,
          float* ssum, float* ssq, float* smn, float* smx, int E) {
  __shared__ __align__(16) __bf16 As[TPB * 16 * 128];   // 32 KB
  __shared__ __align__(16) int nidD[TPB * 16];          // dst node ids (gather)
  __shared__ __align__(16) int nidS[TPB * 16];          // src node ids (gather)
  __shared__ __align__(16) unsigned ndoff[TPB * 16];    // dst*SPAD (pre-scaled)
  const int tid  = threadIdx.x;
  const int nthr = blockDim.x;                          // NT * 32
  const int base = blockIdx.x * (TPB * 16);             // E % (TPB*16) == 0

  for (int j = tid; j < TPB * 16; j += nthr) {
    int d = dst[base + j];
    nidD[j]  = d;
    ndoff[j] = (unsigned)d * SPAD;
    nidS[j]  = src[base + j];
  }
  __syncthreads();

  // async-gather A rows into LDS: TPB*16 edges x [dst row 128B | src row 128B]
  // = TPB*16*16 chunks of 16B via GLOBAL_LOAD_ASYNC_TO_LDS_B128 (ASYNCcnt)
  {
    unsigned lds_base = (unsigned)(unsigned long long)&As[0];
    for (int j = tid; j < TPB * 16 * 16; j += nthr) {
      int e = j >> 4, r = j & 15, part = r >> 3, q = r & 7;
      int node = part ? nidS[e] : nidD[e];
      unsigned lds_addr = lds_base + (unsigned)(e * 256 + part * 128 + q * 16);
      unsigned long long ga =
          (unsigned long long)(hbf + (size_t)node * 64 + q * 8);
      asm volatile("global_load_async_to_lds_b128 %0, %1, off"
                   :: "v"(lds_addr), "v"(ga) : "memory");
    }
    asm volatile("s_wait_asynccnt 0x0" ::: "memory");
  }
  __syncthreads();

  const int lane  = tid & 31;
  const int wave  = tid >> 5;             // n-tile owned by this wave
  const int half  = lane >> 4;
  const int rowm  = lane & 15;
  const int kb    = half * 8;             // A K interleave base
  const int koffB = half * 8;             // B K interleave base (same as A)
  const unsigned ncol = (unsigned)(wave * 16 + rowm);

  // B fragments: loaded once per wave, register-resident for the whole block
  ABfrag bfr[4];
  {
    const __bf16* wp = WpT + (size_t)ncol * 128;
#pragma unroll
    for (int kk = 0; kk < 4; ++kk) {
      *(uint4*)&bfr[kk].u[0] = *(const uint4*)(wp + kk * 32 + koffB);
      *(uint4*)&bfr[kk].u[4] = *(const uint4*)(wp + kk * 32 + 16 + koffB);
    }
  }
  const float bias = bpack[ncol];

  for (int tile = 0; tile < TPB; ++tile) {
    // A fragments from LDS (ISA 16-bit A layout; each half-frag = 16B)
    const __bf16* at = As + tile * 16 * 128 + rowm * 128;
    ABfrag afr[4];
#pragma unroll
    for (int kk = 0; kk < 4; ++kk) {
      *(uint4*)&afr[kk].u[0] = *(const uint4*)(at + kk * 32 + kb);
      *(uint4*)&afr[kk].u[4] = *(const uint4*)(at + kk * 32 + 16 + kb);
    }
    // pre-scaled stat offsets of the 8 C rows this lane holds (M = j+8*half)
    unsigned nd[8];
    *(uint4*)&nd[0] = *(const uint4*)&ndoff[tile * 16 + 8 * half];
    *(uint4*)&nd[4] = *(const uint4*)&ndoff[tile * 16 + 8 * half + 4];

    v8f c = {};
    c = __builtin_amdgcn_wmma_f32_16x16x32_bf16(false, afr[0].v, false, bfr[0].v,
                                                (short)0, c, false, false);
    c = __builtin_amdgcn_wmma_f32_16x16x32_bf16(false, afr[1].v, false, bfr[1].v,
                                                (short)0, c, false, false);
    c = __builtin_amdgcn_wmma_f32_16x16x32_bf16(false, afr[2].v, false, bfr[2].v,
                                                (short)0, c, false, false);
    c = __builtin_amdgcn_wmma_f32_16x16x32_bf16(false, afr[3].v, false, bfr[3].v,
                                                (short)0, c, false, false);

#pragma unroll
    for (int j = 0; j < 8; ++j) {
      float m = c[j] + bias;
      size_t o = (size_t)(nd[j] + ncol);     // zero-extend, no re-scale needed
      atomicAdd(ssum + o, m);
      atomicAdd(ssq + o, m * m);
      atomicMinF(smn + o, m);
      atomicMaxF(smx + o, m);
    }
  }
}

// ---------------- aggregation finalize + W_post + W_lin --------------------
// one 64-thread block per node; generic over (T, Fin, Fo)
__global__ void __launch_bounds__(64)
finalize_post(const float* __restrict__ x,
              const float* __restrict__ ssum, const float* __restrict__ ssq,
              const float* __restrict__ smn,  const float* __restrict__ smx,
              const float* __restrict__ deg,  const float* __restrict__ logd,
              const float* __restrict__ avgp,
              const float* __restrict__ Wpost, const float* __restrict__ bpost,
              const float* __restrict__ Wlin,  const float* __restrict__ blin,
              float* __restrict__ y, int N, int T, int Fin, int Fo) {
  __shared__ float agg[1000];             // [T][4*Fin], max 5*200
  __shared__ float xv[64];
  __shared__ float y1[64];
  const int node = blockIdx.x;
  const int tid  = threadIdx.x;
  const int outc = T * Fo;

  const float dg    = deg[node];
  const float denom = fmaxf(dg, 1.0f);
  const bool  has   = dg > 0.0f;
  const float avgl  = avgp[0];
  const float ld    = logd[node];
  const float amp   = ld / avgl;
  const float att   = avgl / ld;

  const size_t sb = (size_t)node * SPAD;
  for (int idx = tid; idx < T * Fin; idx += 64) {
    int t = idx / Fin, f = idx % Fin;
    float s  = ssum[sb + idx];
    float s2 = ssq[sb + idx];
    float mean = s / denom;
    float var  = fmaxf(s2 / denom - mean * mean, 0.0f);
    float sd   = sqrtf(var + EPS_STD);
    float vmn  = has ? smn[sb + idx] : 0.0f;
    float vmx  = has ? smx[sb + idx] : 0.0f;
    float* ap = agg + t * 4 * Fin;        // order: mean, min, max, std
    ap[f] = mean; ap[Fin + f] = vmn; ap[2 * Fin + f] = vmx; ap[3 * Fin + f] = sd;
  }
  for (int f = tid; f < Fin; f += 64) xv[f] = x[(size_t)node * Fin + f];
  __syncthreads();

  for (int o = tid; o < outc; o += 64) {
    int t = o / Fo, fo = o % Fo;
    const float* W = Wpost + (size_t)t * 13 * Fin * Fo + fo;   // row stride Fo
    float acc = bpost[o];
    for (int c = 0; c < Fin; ++c) acc += xv[c] * W[c * Fo];
    const float* ap = agg + t * 4 * Fin;
    const float* W1 = W + Fin * Fo;       // start of scaled section
    for (int q = 0; q < 4 * Fin; ++q) {
      float a = ap[q];
      acc += a * (W1[q * Fo] + amp * W1[(4 * Fin + q) * Fo]
                             + att * W1[(8 * Fin + q) * Fo]);
    }
    y1[o] = acc;
  }
  __syncthreads();

  for (int o = tid; o < outc; o += 64) {
    float acc = blin[o];
    for (int c = 0; c < outc; ++c) acc += y1[c] * Wlin[(size_t)c * outc + o];
    y[(size_t)node * outc + o] = acc;
  }
}

// ---------------- GRU cell (inp = h_prev, hidden = y) ----------------------
__global__ void gru_kernel(const float* __restrict__ hprev,
                           const float* __restrict__ y,
                           const float* __restrict__ Wih, const float* __restrict__ Whh,
                           const float* __restrict__ bih, const float* __restrict__ bhh,
                           float* __restrict__ hout, int N, int din, int H) {
  int idx = blockIdx.x * blockDim.x + threadIdx.x;
  if (idx >= N * H) return;
  int n = idx / H, j = idx % H;
  const float* hp = hprev + (size_t)n * din;
  const float* yp = y + (size_t)n * H;
  float ir = bih[j], iz = bih[H + j], in_ = bih[2 * H + j];
  for (int c = 0; c < din; ++c) {
    float v = hp[c];
    const float* w = Wih + (size_t)c * 3 * H;
    ir += v * w[j]; iz += v * w[H + j]; in_ += v * w[2 * H + j];
  }
  float hr = bhh[j], hz = bhh[H + j], hn = bhh[2 * H + j];
  for (int c = 0; c < H; ++c) {
    float v = yp[c];
    const float* w = Whh + (size_t)c * 3 * H;
    hr += v * w[j]; hz += v * w[H + j]; hn += v * w[2 * H + j];
  }
  float r  = 1.0f / (1.0f + expf(-(ir + hr)));
  float z  = 1.0f / (1.0f + expf(-(iz + hz)));
  float ns = tanhf(in_ + r * hn);
  hout[idx] = (1.0f - z) * ns + z * yp[j];
}

// ---------------- training-mode BatchNorm over node dim + ReLU -------------
__global__ void bn_stats(const float* __restrict__ h, float* mv, int N, int C) {
  __shared__ float ls[256], ls2[256];
  int c = blockIdx.x;
  float s = 0.0f, s2 = 0.0f;
  for (int n = threadIdx.x; n < N; n += blockDim.x) {
    float v = h[(size_t)n * C + c];
    s += v; s2 += v * v;
  }
  ls[threadIdx.x] = s; ls2[threadIdx.x] = s2;
  __syncthreads();
  for (int off = 128; off > 0; off >>= 1) {
    if (threadIdx.x < off) {
      ls[threadIdx.x]  += ls[threadIdx.x + off];
      ls2[threadIdx.x] += ls2[threadIdx.x + off];
    }
    __syncthreads();
  }
  if (threadIdx.x == 0) {
    float mu = ls[0] / (float)N;
    mv[2 * c]     = mu;
    mv[2 * c + 1] = ls2[0] / (float)N - mu * mu;   // biased variance
  }
}

// BN + ReLU, fused with bf16 repack of the node features for the next conv
// (padding columns of hbf are zeroed once at launch start and never touched)
__global__ void bn_apply_pack(float* h, __bf16* hbf, const float* __restrict__ mv,
                              const float* __restrict__ gamma,
                              const float* __restrict__ beta, int N, int C) {
  int i = blockIdx.x * blockDim.x + threadIdx.x;
  if (i >= N * C) return;
  int n = i / C, c = i % C;
  float v = gamma[c] * (h[i] - mv[2 * c]) * rsqrtf(mv[2 * c + 1] + EPS_BN) + beta[c];
  v = fmaxf(v, 0.0f);
  h[i] = v;
  hbf[(size_t)n * 64 + c] = (__bf16)v;
}

// ===========================================================================
extern "C" void kernel_launch(void* const* d_in, const int* in_sizes, int n_in,
                              void* d_out, int out_size, void* d_ws, size_t ws_size,
                              hipStream_t stream) {
  const int N = 20000, E = 320000, H = 50, NLAYERS = 14;
  const float* x   = (const float*)d_in[0];
  const int*   ei  = (const int*)d_in[1];
  const int*   src = ei;          // edge_index[0]
  const int*   dst = ei + E;      // edge_index[1]

  // params flattened in setup_inputs() insertion order, 12 arrays / layer:
  // 0 W_pre 1 b_pre 2 W_post 3 b_post 4 W_lin 5 b_lin
  // 6 W_ih  7 W_hh  8 b_ih   9 b_hh  10 gamma 11 beta ; readout conv at end
  auto P  = [&](int layer, int off) { return (const float*)d_in[2 + layer * 12 + off]; };
  const int RO = 2 + NLAYERS * 12;
  auto PR = [&](int off) { return (const float*)d_in[RO + off]; };

  // ---------------- workspace bump allocator ----------------
  char* wsp = (char*)d_ws;
  auto alloc = [&](size_t bytes) {
    void* p = wsp; wsp += (bytes + 255) & ~(size_t)255; return p;
  };
  float* deg    = (float*)alloc((size_t)N * 4);
  float* logd   = (float*)alloc((size_t)N * 4);
  float* sumlog = (float*)alloc(256);
  float* avgl   = (float*)alloc(256);
  float* hA     = (float*)alloc((size_t)N * H * 4);
  float* hB     = (float*)alloc((size_t)N * H * 4);
  float* yb     = (float*)alloc((size_t)N * H * 4);
  __bf16* hbf   = (__bf16*)alloc((size_t)N * 64 * 2);
  __bf16* WpT   = (__bf16*)alloc((size_t)256 * 128 * 2);
  float* bpack  = (float*)alloc(256 * 4);
  size_t statN  = (size_t)N * SPAD;
  float* ssum   = (float*)alloc(statN * 4);
  float* ssq    = (float*)alloc(statN * 4);
  float* smn    = (float*)alloc(statN * 4);
  float* smx    = (float*)alloc(statN * 4);
  float* bnmv   = (float*)alloc(2 * H * 4);

  auto fill = [&](float* p, float v, long long cnt) {
    int blocks = (int)((cnt + 255) / 256); if (blocks > 4096) blocks = 4096;
    fill_kernel<<<blocks, 256, 0, stream>>>(p, v, cnt);
  };
  auto init_stats = [&]() {
    init_stats_kernel<<<4096, 256, 0, stream>>>(ssum, ssq, smn, smx, (long long)statN);
  };

  const int EB = E / (TPB * 16);            // 2500 blocks, exact

  // ---------------- one-time zeroing (hbf padding cols stay 0) -------------
  fill((float*)hbf, 0.0f, (long long)N * 64 / 2);
  fill(deg, 0.0f, N);
  fill(sumlog, 0.0f, 1);
  deg_kernel<<<(E + 255) / 256, 256, 0, stream>>>(dst, deg, E);
  logd_kernel<<<256, 256, 0, stream>>>(deg, logd, sumlog, N);
  avg_kernel<<<1, 64, 0, stream>>>(sumlog, avgl, N);

  // ---------------- layer 0 (F_in=2, T=1, Fo=50) ----------------
  init_stats();
  edge_small<<<(E + 255) / 256, 256, 0, stream>>>(x, src, dst, P(0, 0), P(0, 1),
                                                  ssum, ssq, smn, smx, E);
  finalize_post<<<N, 64, 0, stream>>>(x, ssum, ssq, smn, smx, deg, logd, avgl,
                                      P(0, 2), P(0, 3), P(0, 4), P(0, 5),
                                      yb, N, 1, 2, 50);
  gru_kernel<<<(N * H + 255) / 256, 256, 0, stream>>>(x, yb, P(0, 6), P(0, 7),
                                                      P(0, 8), P(0, 9), hA, N, 2, H);
  bn_stats<<<H, 256, 0, stream>>>(hA, bnmv, N, H);
  bn_apply_pack<<<(N * H + 255) / 256, 256, 0, stream>>>(hA, hbf, bnmv,
                                                         P(0, 10), P(0, 11), N, H);

  float* h_cur = hA;
  float* h_nxt = hB;

  // ---------------- layers 1..13 (F_in=50, T=5, Fo=10, NT=16) --------------
  for (int L = 1; L < NLAYERS; ++L) {
    pack_w<<<(256 * 128 + 255) / 256, 256, 0, stream>>>(P(L, 0), P(L, 1), WpT, bpack, 5, H);
    init_stats();
    edge_wmma<<<EB, 16 * 32, 0, stream>>>(hbf, WpT, bpack, src, dst,
                                          ssum, ssq, smn, smx, E);
    finalize_post<<<N, 64, 0, stream>>>(h_cur, ssum, ssq, smn, smx, deg, logd, avgl,
                                        P(L, 2), P(L, 3), P(L, 4), P(L, 5),
                                        yb, N, 5, H, 10);
    gru_kernel<<<(N * H + 255) / 256, 256, 0, stream>>>(h_cur, yb, P(L, 6), P(L, 7),
                                                        P(L, 8), P(L, 9), h_nxt, N, H, H);
    bn_stats<<<H, 256, 0, stream>>>(h_nxt, bnmv, N, H);
    bn_apply_pack<<<(N * H + 255) / 256, 256, 0, stream>>>(h_nxt, hbf, bnmv,
                                                           P(L, 10), P(L, 11), N, H);
    float* t = h_cur; h_cur = h_nxt; h_nxt = t;
  }

  // ---------------- readout PNAConv (F_in=50, T=1, Fo=1, NT=4) -------------
  pack_w<<<(256 * 128 + 255) / 256, 256, 0, stream>>>(PR(0), PR(1), WpT, bpack, 1, H);
  init_stats();
  edge_wmma<<<EB, 4 * 32, 0, stream>>>(hbf, WpT, bpack, src, dst,
                                       ssum, ssq, smn, smx, E);
  finalize_post<<<N, 64, 0, stream>>>(h_cur, ssum, ssq, smn, smx, deg, logd, avgl,
                                      PR(2), PR(3), PR(4), PR(5),
                                      (float*)d_out, N, 1, 50, 1);
}
